// GNNEncoder_43164421325635
// MI455X (gfx1250) — matrix operand order repaired and checked by reading, not compiled, not run on hardware
//
#include <hip/hip_runtime.h>

// ============================================================================
// GatedGCN diffusion forward for MI455X (gfx1250), wave32 + WMMA f16->f32.
//
// Dominant cost: per-layer edge GEMM (131072x128)@(128x128) = 4.3 GFLOP,
// bandwidth floor ~5.8us at 23.3 TB/s => must run on the f16 WMMA pipe
// (fp32 accumulate). fp32 master tensors stay in HBM for BN stats/residuals;
// an f16 mirror of the edge tensor feeds the GEMM A-panels so staging is a
// pure byte copy done with GLOBAL_LOAD_ASYNC_TO_LDS_B128 (ASYNCcnt DMA).
//
// Workspace requirement: ~162 MB.
// ============================================================================

typedef __attribute__((ext_vector_type(16))) _Float16 v16h;
typedef __attribute__((ext_vector_type(8)))  float    v8f;
typedef int v4i __attribute__((__vector_size__(4 * sizeof(int))));

#define DD   128            // feature dim
#define LDA  136            // padded LDS row stride (halves): 272B = 17*16B
                            // (async b128 chunks stay 16B-aligned; 68 banks
                            //  ≡ 4 mod 64 -> 16 fragment rows hit 16 banks)
#define TM   64             // M rows per block in GEMM kernels
#define NROW 512            // B*N
#define EROW 131072L        // B*N*N

#if __has_builtin(__builtin_amdgcn_global_load_async_to_lds_b128)
#define HAS_ASYNC_LDS 1
#else
#define HAS_ASYNC_LDS 0
#endif

__device__ __forceinline__ void async_copy_b128(const void* g, void* l) {
#if HAS_ASYNC_LDS
  __builtin_amdgcn_global_load_async_to_lds_b128((v4i*)g, (v4i*)l, 0, 0);
#else
  (void)g; (void)l;
#endif
}

__device__ __forceinline__ void async_wait_all() {
#if HAS_ASYNC_LDS
#if __has_builtin(__builtin_amdgcn_s_wait_asynccnt)
  __builtin_amdgcn_s_wait_asynccnt(0);
#else
  asm volatile("s_wait_asynccnt 0x0" ::: "memory");
#endif
#endif
}

union FragU { v16h h; unsigned int u[8]; };

// A-fragment: 16x32 f16. Lane L: row M = L%16, half = L/16.
// v16h elem pairs (2i,2i+1) map to K = kc*32 + half*8 + {0..7} for i<4,
// and K = kc*32 + 16 + half*8 + {0..7} for i>=4  (ISA 7.12.2 table).
__device__ inline v16h load_a_frag(const _Float16* s, int row0, int lane, int kc) {
  const _Float16* base = s + (row0 + (lane & 15)) * LDA + kc * 32 + (lane >> 4) * 8;
  FragU f;
#pragma unroll
  for (int vg = 0; vg < 4; ++vg) f.u[vg]     = *(const unsigned int*)(base + 2 * vg);
#pragma unroll
  for (int vg = 0; vg < 4; ++vg) f.u[4 + vg] = *(const unsigned int*)(base + 16 + 2 * vg);
  return f.h;
}

// B-fragment: 32x16 f16 from TRANSPOSED weight LDS (sT[n][k]).
// Lane L: col N = n0 + L%16, K = kc*32 + (L/16)*8 + {0..7}, then +16 (table order, matches A).
__device__ inline v16h load_b_frag(const _Float16* sT, int n0, int lane, int kc) {
  const _Float16* base = sT + (n0 + (lane & 15)) * LDA + kc * 32 + (lane >> 4) * 8;
  FragU f;
#pragma unroll
  for (int i = 0; i < 8; ++i) f.u[i] = *(const unsigned int*)(base + (i >> 2) * 16 + 2 * (i & 3));
  return f.h;
}

// ---------------------------------------------------------------------------
// Node GEMM (fp32 activations): out[M x 128] = A[M x 128] @ W[128 x 128].
// ---------------------------------------------------------------------------
__global__ __launch_bounds__(128) void k_gemm_f16w(
    const float* __restrict__ A, const float* __restrict__ W,
    float* __restrict__ out)
{
  __shared__ _Float16 sW[DD * LDA];   // transposed weights  [n][k]
  __shared__ _Float16 sA[TM * LDA];   // activations         [m][k]
  const int tid = threadIdx.x;
  const long grow0 = (long)blockIdx.x * TM;

  for (int idx = tid; idx < DD * DD; idx += 128) {
    int k = idx >> 7, n = idx & 127;
    sW[n * LDA + k] = (_Float16)W[idx];
  }
  for (int idx = tid; idx < TM * DD; idx += 128) {
    int r = idx >> 7, c = idx & 127;
    sA[r * LDA + c] = (_Float16)A[(grow0 + r) * DD + c];
  }
  __syncthreads();

  const int lane = tid & 31, wv = tid >> 5;
  const int rowb = wv * 16;
  v8f acc[8];
  v8f z = {};
#pragma unroll
  for (int n = 0; n < 8; ++n) acc[n] = z;

#pragma unroll
  for (int kc = 0; kc < 4; ++kc) {
    v16h af = load_a_frag(sA, rowb, lane, kc);
#pragma unroll
    for (int n = 0; n < 8; ++n) {
      v16h bf = load_b_frag(sW, n * 16, lane, kc);
      acc[n] = __builtin_amdgcn_wmma_f32_16x16x32_f16(
          false, af, false, bf, (short)0, acc[n], false, false);
    }
  }

  const int col0 = lane & 15;
  const int mh = (lane >> 4) * 8;
#pragma unroll
  for (int n = 0; n < 8; ++n) {
    const int col = n * 16 + col0;
#pragma unroll
    for (int r = 0; r < 8; ++r) {
      long row = grow0 + rowb + mh + r;
      out[row * DD + col] = acc[n][r];
    }
  }
}

// ---------------------------------------------------------------------------
// Edge GEMM (f16 mirror input, async-DMA staged):
// gate[row,:] = e16[row,:] @ C + Ah[b*N+i,:] + Bh[b*N+j,:], row=b*N*N+i*N+j.
// ---------------------------------------------------------------------------
__global__ __launch_bounds__(128) void k_gemm_e16(
    const _Float16* __restrict__ A16, const float* __restrict__ W,
    float* __restrict__ out,
    const float* __restrict__ addI, const float* __restrict__ addJ)
{
  __shared__ _Float16 sW[DD * LDA];
  __shared__ _Float16 sA[TM * LDA];
  const int tid = threadIdx.x;
  const long grow0 = (long)blockIdx.x * TM;

#if HAS_ASYNC_LDS
  // A panel is already f16: 64 rows x 256B, DMA'd 16B/lane (ASYNCcnt).
  for (int c = tid; c < TM * 16; c += 128) {
    int r = c >> 4, seg = c & 15;
    async_copy_b128(A16 + (grow0 + r) * DD + seg * 8, sA + r * LDA + seg * 8);
  }
#else
  for (int idx = tid; idx < TM * DD; idx += 128) {
    int r = idx >> 7, c = idx & 127;
    sA[r * LDA + c] = A16[(grow0 + r) * DD + c];
  }
#endif
  // Weights need fp32->f16 transpose; overlap with the async DMA above.
  for (int idx = tid; idx < DD * DD; idx += 128) {
    int k = idx >> 7, n = idx & 127;
    sW[n * LDA + k] = (_Float16)W[idx];
  }
  if (blockIdx.x + 1 < gridDim.x)
    __builtin_prefetch(A16 + (grow0 + TM) * DD + tid * 64, 0, 1);
  async_wait_all();
  __syncthreads();

  const int lane = tid & 31, wv = tid >> 5;
  const int rowb = wv * 16;
  v8f acc[8];
  v8f z = {};
#pragma unroll
  for (int n = 0; n < 8; ++n) acc[n] = z;

#pragma unroll
  for (int kc = 0; kc < 4; ++kc) {
    v16h af = load_a_frag(sA, rowb, lane, kc);
#pragma unroll
    for (int n = 0; n < 8; ++n) {
      v16h bf = load_b_frag(sW, n * 16, lane, kc);
      acc[n] = __builtin_amdgcn_wmma_f32_16x16x32_f16(
          false, af, false, bf, (short)0, acc[n], false, false);
    }
  }

  const int col0 = lane & 15;
  const int mh = (lane >> 4) * 8;
#pragma unroll
  for (int n = 0; n < 8; ++n) {
    const int col = n * 16 + col0;
#pragma unroll
    for (int r = 0; r < 8; ++r) {
      long row = grow0 + rowb + mh + r;
      long bi = row >> 8;          // b*N + i
      long b  = row >> 16;         // batch
      long j  = row & 255;
      out[row * DD + col] = acc[n][r]
                          + addI[bi * DD + col]
                          + addJ[(b * 256 + j) * DD + col];
    }
  }
}

// ---------------------------------------------------------------------------
// Output head: v[row] = relu(e16[row,:] @ W1 + b1) @ w2 + b2   (W1: 128x64)
// ---------------------------------------------------------------------------
__global__ __launch_bounds__(128) void k_head(
    const _Float16* __restrict__ e16, const float* __restrict__ w1,
    const float* __restrict__ b1, const float* __restrict__ w2,
    const float* __restrict__ b2, float* __restrict__ out)
{
  __shared__ _Float16 sW[64 * LDA];
  __shared__ _Float16 sA[TM * LDA];
  const int tid = threadIdx.x;
  const long grow0 = (long)blockIdx.x * TM;

#if HAS_ASYNC_LDS
  for (int c = tid; c < TM * 16; c += 128) {
    int r = c >> 4, seg = c & 15;
    async_copy_b128(e16 + (grow0 + r) * DD + seg * 8, sA + r * LDA + seg * 8);
  }
#else
  for (int idx = tid; idx < TM * DD; idx += 128) {
    int r = idx >> 7, c = idx & 127;
    sA[r * LDA + c] = e16[(grow0 + r) * DD + c];
  }
#endif
  for (int idx = tid; idx < DD * 64; idx += 128) {
    int k = idx >> 6, n = idx & 63;
    sW[n * LDA + k] = (_Float16)w1[idx];
  }
  if (blockIdx.x + 1 < gridDim.x)
    __builtin_prefetch(e16 + (grow0 + TM) * DD + tid * 64, 0, 1);
  async_wait_all();
  __syncthreads();

  const int lane = tid & 31, wv = tid >> 5;
  const int rowb = wv * 16, col0 = lane & 15;
  v8f acc[4];
  v8f z = {};
#pragma unroll
  for (int n = 0; n < 4; ++n) acc[n] = z;

#pragma unroll
  for (int kc = 0; kc < 4; ++kc) {
    v16h af = load_a_frag(sA, rowb, lane, kc);
#pragma unroll
    for (int n = 0; n < 4; ++n) {
      v16h bf = load_b_frag(sW, n * 16, lane, kc);
      acc[n] = __builtin_amdgcn_wmma_f32_16x16x32_f16(
          false, af, false, bf, (short)0, acc[n], false, false);
    }
  }

  float b1c[4], w2c[4];
#pragma unroll
  for (int n = 0; n < 4; ++n) { b1c[n] = b1[n * 16 + col0]; w2c[n] = w2[n * 16 + col0]; }

  float p[8];
#pragma unroll
  for (int r = 0; r < 8; ++r) p[r] = 0.f;
#pragma unroll
  for (int n = 0; n < 4; ++n)
#pragma unroll
    for (int r = 0; r < 8; ++r)
      p[r] += fmaxf(acc[n][r] + b1c[n], 0.f) * w2c[n];

  // Reduce the 16 lanes that share a row (width-16 butterfly).
#pragma unroll
  for (int r = 0; r < 8; ++r) {
    p[r] += __shfl_xor(p[r], 8, 16);
    p[r] += __shfl_xor(p[r], 4, 16);
    p[r] += __shfl_xor(p[r], 2, 16);
    p[r] += __shfl_xor(p[r], 1, 16);
  }
  if (col0 == 0) {
    const float bb = b2[0];
    const int mh = (lane >> 4) * 8;
#pragma unroll
    for (int r = 0; r < 8; ++r) out[grow0 + rowb + mh + r] = p[r] + bb;
  }
}

// ---------------------------------------------------------------------------
// Elementwise / reduction kernels (bandwidth-bound)
// ---------------------------------------------------------------------------
__global__ void k_zero(float* __restrict__ p, int n) {
  int i = blockIdx.x * blockDim.x + threadIdx.x;
  if (i < n) p[i] = 0.f;
}

// t_feat[b,:] = sin_emb(t_b) @ time_w + time_b
__global__ __launch_bounds__(128) void k_time(
    const float* __restrict__ t, const float* __restrict__ tw,
    const float* __restrict__ tb, float* __restrict__ tfeat)
{
  __shared__ float emb[DD];
  const int d = threadIdx.x, b = blockIdx.x;
  const float tv = t[b];
  const float fr = __expf(-logf(10000.f) * (float)(d & 63) / 64.f);
  const float arg = tv * fr;
  emb[d] = (d < 64) ? sinf(arg) : cosf(arg);
  __syncthreads();
  float s = tb[d];
  for (int k = 0; k < DD; ++k) s += emb[k] * tw[k * DD + d];
  tfeat[b * DD + d] = s;
}

// h[row,d] = relu(coords @ node_w + node_b) + t_feat[b,d]
__global__ void k_node_embed(
    const float* __restrict__ coords, const float* __restrict__ nw,
    const float* __restrict__ nb, const float* __restrict__ tfeat,
    float* __restrict__ h)
{
  int idx = blockIdx.x * blockDim.x + threadIdx.x;   // NROW*128
  int d = idx & 127, row = idx >> 7, b = row >> 8;
  float x = coords[row * 2], y = coords[row * 2 + 1];
  float v = fmaxf(x * nw[d] + y * nw[DD + d] + nb[d], 0.f);
  h[idx] = v + tfeat[b * DD + d];
}

// e[b,i,j,d] = relu([dist, adj] @ edge_w + edge_b)   (fp32 master + f16 mirror)
__global__ void k_edge_embed(
    const float* __restrict__ coords, const float* __restrict__ adj,
    const float* __restrict__ ew, const float* __restrict__ eb,
    float* __restrict__ e, _Float16* __restrict__ e16)
{
  long idx = (long)blockIdx.x * blockDim.x + threadIdx.x;  // EROW*128
  int d = (int)(idx & 127);
  long r = idx >> 7;
  int j = (int)(r & 255), i = (int)((r >> 8) & 255), b = (int)(r >> 16);
  float xi = coords[(b * 256 + i) * 2],     yi = coords[(b * 256 + i) * 2 + 1];
  float xj = coords[(b * 256 + j) * 2],     yj = coords[(b * 256 + j) * 2 + 1];
  float dx = xi - xj, dy = yi - yj;
  float dist = sqrtf(dx * dx + dy * dy + 1e-12f);
  float a = adj[r];
  float v = fmaxf(dist * ew[d] + a * ew[DD + d] + eb[d], 0.f);
  e[idx] = v;
  e16[idx] = (_Float16)v;
}

// Per-channel partial sums (mean/var) via atomics: st[0..127]=sum, [128..255]=sumsq
__global__ __launch_bounds__(128) void k_stats(
    const float* __restrict__ x, int rowsPerBlk, float* __restrict__ st)
{
  const int d = threadIdx.x;
  const long r0 = (long)blockIdx.x * rowsPerBlk;
  float s = 0.f, s2 = 0.f;
  for (int r = 0; r < rowsPerBlk; ++r) {
    float v = x[(r0 + r) * DD + d];
    s += v; s2 += v * v;
  }
  atomicAdd(&st[d], s);
  atomicAdd(&st[DD + d], s2);
}

__global__ __launch_bounds__(128) void k_finalize(
    const float* __restrict__ st, float n, float* __restrict__ mi)
{
  int d = threadIdx.x;
  float m = st[d] / n;
  float var = st[DD + d] / n - m * m;           // biased var (training-mode BN)
  mi[d] = m;
  mi[DD + d] = rsqrtf(var + 1e-5f);
}

// Fused per (b,i): agg_d = sum_j sigmoid(gate)*Vh[b,j,d];
//                  e[b,i,j,d] = relu(BN(gate)) + e   (in place, + f16 mirror);
//                  hpre[b,i,d] = relu(Uh + agg)
__global__ __launch_bounds__(128) void k_agg_bn(
    const float* __restrict__ gate, const float* __restrict__ Vh,
    const float* __restrict__ Uh, const float* __restrict__ mi,
    const float* __restrict__ ge, const float* __restrict__ be,
    float* __restrict__ e, _Float16* __restrict__ e16,
    float* __restrict__ hpre)
{
  const int d = threadIdx.x;
  const int bi = blockIdx.x;          // b*N + i
  const int b = bi >> 8;
  const float m = mi[d], inv = mi[DD + d];
  const float g = ge[d], bb = be[d];
  float acc = 0.f;
  const long base = (long)bi * 256;
  for (int j = 0; j < 256; ++j) {
    long idx = (base + j) * DD + d;
    float gi = gate[idx];
    acc += (1.f / (1.f + __expf(-gi))) * Vh[(b * 256 + j) * DD + d];
    float bn = (gi - m) * inv * g + bb;
    float v = fmaxf(bn, 0.f) + e[idx];
    e[idx] = v;
    e16[idx] = (_Float16)v;
  }
  hpre[bi * DD + d] = fmaxf(Uh[bi * DD + d] + acc, 0.f);
}

// h[row,d] = BN(hpre)*g_n + b_n + h[row,d]
__global__ void k_node_apply(
    const float* __restrict__ hpre, const float* __restrict__ mi,
    const float* __restrict__ gn, const float* __restrict__ bn_,
    float* __restrict__ h)
{
  int idx = blockIdx.x * blockDim.x + threadIdx.x;
  int d = idx & 127;
  float v = (hpre[idx] - mi[d]) * mi[DD + d] * gn[d] + bn_[d];
  h[idx] = v + h[idx];
}

// ============================================================================
// Host-side orchestration
// ============================================================================
extern "C" void kernel_launch(void* const* d_in, const int* in_sizes, int n_in,
                              void* d_out, int out_size, void* d_ws, size_t ws_size,
                              hipStream_t stream) {
  (void)in_sizes; (void)n_in; (void)out_size; (void)ws_size;
  // Inputs in setup_inputs() insertion order, params flattened depth-first:
  const float* coords = (const float*)d_in[0];   // (2,256,2)
  const float* adj    = (const float*)d_in[1];   // (2,256,256)
  const float* tt     = (const float*)d_in[2];   // (2,)
  const float* node_w = (const float*)d_in[3];
  const float* node_b = (const float*)d_in[4];
  const float* edge_w = (const float*)d_in[5];
  const float* edge_b = (const float*)d_in[6];
  const float* time_w = (const float*)d_in[7];
  const float* time_b = (const float*)d_in[8];
  const float* out_w1 = (const float*)d_in[9];
  const float* out_b1 = (const float*)d_in[10];
  const float* out_w2 = (const float*)d_in[11];
  const float* out_b2 = (const float*)d_in[12];
  // layers at 13 + 9*l: A, Bm, C, U, V, g_n, b_n, g_e, b_e

  float* ws = (float*)d_ws;
  float* h     = ws;                 // 65536
  float* hpre  = h + 65536;          // 65536
  float* tfeat = hpre + 65536;       // 256
  float* Ah    = tfeat + 256;        // 65536
  float* Bh    = Ah + 65536;         // 65536
  float* Vh    = Bh + 65536;         // 65536
  float* Uh    = Vh + 65536;         // 65536
  float* st    = Uh + 65536;         // 256
  float* mi    = st + 256;           // 256
  float* e     = mi + 256;           // 16777216 (64 MB)
  float* gate  = e + 16777216;       // 16777216 (64 MB)
  _Float16* e16 = (_Float16*)(gate + 16777216);  // 16777216 halves (32 MB)

  // ---- embeddings ----
  k_time<<<2, 128, 0, stream>>>(tt, time_w, time_b, tfeat);
  k_node_embed<<<(NROW * DD) / 256, 256, 0, stream>>>(coords, node_w, node_b, tfeat, h);
  k_edge_embed<<<(unsigned)((EROW * DD) / 256), 256, 0, stream>>>(coords, adj, edge_w, edge_b, e, e16);

  // ---- 6 GatedGCN layers ----
  for (int l = 0; l < 6; ++l) {
    const float* Wa = (const float*)d_in[13 + 9 * l + 0];
    const float* Wb = (const float*)d_in[13 + 9 * l + 1];
    const float* Wc = (const float*)d_in[13 + 9 * l + 2];
    const float* Wu = (const float*)d_in[13 + 9 * l + 3];
    const float* Wv = (const float*)d_in[13 + 9 * l + 4];
    const float* g_n = (const float*)d_in[13 + 9 * l + 5];
    const float* b_n = (const float*)d_in[13 + 9 * l + 6];
    const float* g_e = (const float*)d_in[13 + 9 * l + 7];
    const float* b_e = (const float*)d_in[13 + 9 * l + 8];

    // node projections (512x128 @ 128x128), WMMA
    k_gemm_f16w<<<NROW / TM, 128, 0, stream>>>(h, Wa, Ah);
    k_gemm_f16w<<<NROW / TM, 128, 0, stream>>>(h, Wb, Bh);
    k_gemm_f16w<<<NROW / TM, 128, 0, stream>>>(h, Wv, Vh);
    k_gemm_f16w<<<NROW / TM, 128, 0, stream>>>(h, Wu, Uh);

    // gate_input = e@C + Ah[b,i,:] + Bh[b,j,:]  (WMMA + async-LDS staging)
    k_gemm_e16<<<(unsigned)(EROW / TM), 128, 0, stream>>>(e16, Wc, gate, Ah, Bh);

    // edge BatchNorm stats over 131072 rows
    k_zero<<<1, 256, 0, stream>>>(st, 256);
    k_stats<<<512, 128, 0, stream>>>(gate, 256, st);
    k_finalize<<<1, 128, 0, stream>>>(st, 131072.f, mi);

    // fused: gated aggregation + edge BN apply (+relu, +residual, in place)
    k_agg_bn<<<NROW, 128, 0, stream>>>(gate, Vh, Uh, mi, g_e, b_e, e, e16, hpre);

    // node BatchNorm stats over 512 rows
    k_zero<<<1, 256, 0, stream>>>(st, 256);
    k_stats<<<1, 128, 0, stream>>>(hpre, 512, st);
    k_finalize<<<1, 128, 0, stream>>>(st, 512.f, mi);
    k_node_apply<<<(NROW * DD) / 256, 256, 0, stream>>>(hpre, mi, g_n, b_n, h);
  }

  // ---- output head (WMMA, 131072x128 @ 128x64, fused relu·w2 reduce) ----
  k_head<<<(unsigned)(EROW / TM), 128, 0, stream>>>(e16, out_w1, out_b1, out_w2, out_b2,
                                                    (float*)d_out);
}
